// EfficientFusionModule_80968723464689
// MI455X (gfx1250) — compile-verified, matrix operand order checked by reading, not run on hardware
//
#include <hip/hip_runtime.h>
#include <hip/hip_bf16.h>
#include <math.h>

#define D_MODEL 128
#define D_INNER 256
#define D_STATE 16
#define DT_RANK 8
#define BATCH 2
#define HH 96
#define WW 96
#define LSEQ (HH * WW)          // 9216
#define BL (BATCH * LSEQ)       // 18432
#define NPROJ (DT_RANK + 2 * D_STATE)  // 40
#define NPROJ_PAD 48

typedef __attribute__((ext_vector_type(16))) __bf16 v16bf;
typedef __attribute__((ext_vector_type(8)))  float  v8f;

// Load 8 consecutive f32, convert to bf16, place in halves [base, base+8) of v.
__device__ __forceinline__ void fill8(v16bf& v, const int base, const float* __restrict__ p) {
    const float4* q = reinterpret_cast<const float4*>(p);
    float4 f0 = q[0];
    float4 f1 = q[1];
    v[base + 0] = (__bf16)f0.x; v[base + 1] = (__bf16)f0.y;
    v[base + 2] = (__bf16)f0.z; v[base + 3] = (__bf16)f0.w;
    v[base + 4] = (__bf16)f1.x; v[base + 5] = (__bf16)f1.y;
    v[base + 6] = (__bf16)f1.z; v[base + 7] = (__bf16)f1.w;
}

// D(M x N) = A(M x K) * W(N x K)^T with v_wmma_f32_16x16x32_bf16.
// One wave computes a 16 x (NT*16) tile: A operand loaded once per K-step,
// reused across NT independent WMMAs (hides the 4-coexec BF16 WAR hazard).
// K = KTILES*32, fully unrolled. W must have >= Ngroups*NT*16 valid rows
// (caller zero-pads). Store: columns >= Nvalid skipped; if out1 != nullptr,
// columns >= split go to out1 at (column - split).
template <int KTILES, int NT>
__global__ void gemm_bf16_wmma(const float* __restrict__ A,
                               const float* __restrict__ W,
                               float* __restrict__ out0,
                               float* __restrict__ out1,
                               int Mtiles, int Ngroups, int Nvalid,
                               int split, int ld0, int ld1)
{
    constexpr int K = KTILES * 32;
    const int wave  = (blockIdx.x * blockDim.x + threadIdx.x) >> 5;
    if (wave >= Mtiles * Ngroups) return;      // wave-uniform (grids exact)

    const int mt = wave / Ngroups;
    const int ng = wave % Ngroups;
    const int lane  = threadIdx.x & 31;
    const int rowl  = lane & 15;               // A row / B column for this lane
    const int hi    = lane >> 4;               // lane half -> K sub-block
    const int kbase = hi * 8;
    const int m0 = mt * 16;
    const int n0 = ng * NT * 16;

    const float* arow = A + (size_t)(m0 + rowl) * K;
    const float* brow[NT];
    #pragma unroll
    for (int j = 0; j < NT; ++j)
        brow[j] = W + (size_t)(n0 + j * 16 + rowl) * K;

    v8f acc[NT];
    #pragma unroll
    for (int j = 0; j < NT; ++j)
        #pragma unroll
        for (int r = 0; r < 8; ++r) acc[j][r] = 0.f;

    #pragma unroll
    for (int kt = 0; kt < KTILES; ++kt) {
        const int k0 = kt * 32;
        v16bf a;
        // 16-bit A 16x32 layout: halves 0..7 -> K=k0+kbase..+7,
        //                        halves 8..15 -> K=k0+16+kbase..+7
        fill8(a, 0, arow + k0 + kbase);
        fill8(a, 8, arow + k0 + 16 + kbase);
        #pragma unroll
        for (int j = 0; j < NT; ++j) {
            v16bf b;
            fill8(b, 0, brow[j] + k0 + kbase);
            fill8(b, 8, brow[j] + k0 + 16 + kbase);
            acc[j] = __builtin_amdgcn_wmma_f32_16x16x32_bf16(
                         false, a, false, b, (short)0, acc[j], false, false);
        }
    }

    #pragma unroll
    for (int j = 0; j < NT; ++j) {
        const int nglob = n0 + j * 16 + rowl;  // C/D: N = lane&15
        if (nglob < Nvalid) {
            #pragma unroll
            for (int r = 0; r < 8; ++r) {      // C/D: VGPR r -> M = r + 8*hi
                const int m = m0 + r + hi * 8;
                const float v = acc[j][r];
                if (out1 != nullptr && nglob >= split)
                    out1[(size_t)m * ld1 + (nglob - split)] = v;
                else
                    out0[(size_t)m * ld0 + nglob] = v;
            }
        }
    }
}

// Zero-pad x_proj_w (40,256) -> (48,256) so the GEMM loads are unguarded.
__global__ void pad_xproj_kernel(const float* __restrict__ src, float* __restrict__ dst)
{
    const int t = blockIdx.x * blockDim.x + threadIdx.x;
    if (t >= NPROJ_PAD * D_INNER) return;
    const int row = t >> 8;
    dst[t] = (row < NPROJ) ? src[t] : 0.f;
}

// Depthwise 3x3 'SAME' conv + bias + SiLU.  xi,xs: (B, H, W, D_INNER)
__global__ void conv_silu_kernel(const float* __restrict__ xi,
                                 const float* __restrict__ cw,
                                 const float* __restrict__ cb,
                                 float* __restrict__ xs)
{
    const int t = blockIdx.x * blockDim.x + threadIdx.x;
    if (t >= BL * D_INNER) return;
    const int d   = t & (D_INNER - 1);
    const int pos = t >> 8;
    const int w   = pos % WW;
    const int hb  = pos / WW;
    const int h   = hb % HH;
    const int b   = hb / HH;

    float sum = cb[d];
    #pragma unroll
    for (int dh = -1; dh <= 1; ++dh) {
        const int hh = h + dh;
        if (hh < 0 || hh >= HH) continue;
        #pragma unroll
        for (int dw = -1; dw <= 1; ++dw) {
            const int ww2 = w + dw;
            if (ww2 < 0 || ww2 >= WW) continue;
            sum += xi[(((size_t)b * HH + hh) * WW + ww2) * D_INNER + d]
                 * cw[d * 9 + (dh + 1) * 3 + (dw + 1)];
        }
    }
    xs[t] = sum / (1.0f + __expf(-sum));       // SiLU
}

// delta = softplus(dts @ dt_w^T + dt_b); K = DT_RANK = 8 (pure VALU).
__global__ void delta_kernel(const float* __restrict__ xdbl,
                             const float* __restrict__ dtw,
                             const float* __restrict__ dtb,
                             float* __restrict__ delta)
{
    const int t = blockIdx.x * blockDim.x + threadIdx.x;
    if (t >= BL * D_INNER) return;
    const int d = t & (D_INNER - 1);
    const int m = t >> 8;
    const float* row = xdbl + (size_t)m * NPROJ;
    float acc = dtb[d];
    #pragma unroll
    for (int r = 0; r < DT_RANK; ++r)
        acc += row[r] * dtw[d * DT_RANK + r];
    delta[t] = (acc > 20.f) ? acc : log1pf(__expf(acc));
}

// Selective scan: one 16-lane group per (b,d) sequence, one state n per lane.
// 512 sequences -> 256 waves; launched 128 blocks x 64 threads so ~128 WGPs
// each carry 2 waves (latency overlap on the serial recurrence).
__global__ void scan_kernel(const float* __restrict__ xs,
                            const float* __restrict__ delta,
                            const float* __restrict__ xdbl,
                            const float* __restrict__ A_logs,
                            const float* __restrict__ Ds,
                            float* __restrict__ y)
{
    const int wave = (blockIdx.x * blockDim.x + threadIdx.x) >> 5;
    const int lane = threadIdx.x & 31;
    const int g    = lane >> 4;                // which of 2 sequences in wave
    const int n    = lane & 15;                // state index
    const int ch   = wave * 2 + g;             // 0..511
    const int b    = ch >> 8;
    const int d    = ch & (D_INNER - 1);

    const float Adn = -__expf(A_logs[d * D_STATE + n]);
    const float Dsd = Ds[d];
    float h = 0.f;

    for (int l = 0; l < LSEQ; ++l) {
        const size_t m  = (size_t)b * LSEQ + l;
        const float dt  = delta[m * D_INNER + d];
        const float xv  = xs[m * D_INNER + d];
        const float Bv  = xdbl[m * NPROJ + DT_RANK + n];
        const float Cv  = xdbl[m * NPROJ + DT_RANK + D_STATE + n];
        const float dA  = __expf(dt * Adn);
        h = dA * h + (dt * xv) * Bv;
        float p = h * Cv;
        // reduce over the 16 lanes of this group (xor masks <16 stay in-half)
        p += __shfl_xor(p, 1, 32);
        p += __shfl_xor(p, 2, 32);
        p += __shfl_xor(p, 4, 32);
        p += __shfl_xor(p, 8, 32);
        if (n == 0) y[m * D_INNER + d] = p + Dsd * xv;
    }
}

// Per-position LayerNorm over 256 channels, then * SiLU(z).
__global__ void ln_gate_kernel(const float* __restrict__ y,
                               const float* __restrict__ z,
                               const float* __restrict__ g,
                               const float* __restrict__ be,
                               float* __restrict__ yg)
{
    __shared__ float sbuf[D_INNER];
    const int m = blockIdx.x;
    const int d = threadIdx.x;
    const float v = y[(size_t)m * D_INNER + d];

    sbuf[d] = v; __syncthreads();
    for (int s = 128; s > 0; s >>= 1) {
        if (d < s) sbuf[d] += sbuf[d + s];
        __syncthreads();
    }
    const float mu = sbuf[0] * (1.0f / D_INNER);
    __syncthreads();

    const float dv = v - mu;
    sbuf[d] = dv * dv; __syncthreads();
    for (int s = 128; s > 0; s >>= 1) {
        if (d < s) sbuf[d] += sbuf[d + s];
        __syncthreads();
    }
    const float var = sbuf[0] * (1.0f / D_INNER);

    const float nrm = dv * rsqrtf(var + 1e-5f) * g[d] + be[d];
    const float zv  = z[(size_t)m * D_INNER + d];
    yg[(size_t)m * D_INNER + d] = nrm * (zv / (1.0f + __expf(-zv)));
}

extern "C" void kernel_launch(void* const* d_in, const int* in_sizes, int n_in,
                              void* d_out, int out_size, void* d_ws, size_t ws_size,
                              hipStream_t stream)
{
    const float* x        = (const float*)d_in[0];   // (2,96,96,128)
    const float* in_proj  = (const float*)d_in[1];   // (512,128)
    const float* conv_w   = (const float*)d_in[2];   // (256,1,3,3)
    const float* conv_b   = (const float*)d_in[3];   // (256,)
    const float* x_proj_w = (const float*)d_in[4];   // (1,40,256)
    const float* dt_w     = (const float*)d_in[5];   // (1,256,8)
    const float* dt_b     = (const float*)d_in[6];   // (1,256)
    const float* A_logs   = (const float*)d_in[7];   // (256,16)
    const float* Ds       = (const float*)d_in[8];   // (256,)
    const float* ln_g     = (const float*)d_in[9];   // (256,)
    const float* ln_b     = (const float*)d_in[10];  // (256,)
    const float* out_proj = (const float*)d_in[11];  // (128,256)

    float* ws = (float*)d_ws;
    const size_t C = (size_t)BL * D_INNER;
    float* xi    = ws;               // (BL,256) pre-conv, reused as gated output
    float* z     = xi + C;           // (BL,256)
    float* xs    = z + C;            // (BL,256) post conv+silu
    float* delta = xs + C;           // (BL,256)
    float* ybuf  = delta + C;        // (BL,256)
    float* xdbl  = ybuf + C;         // (BL,40)
    float* wpad  = xdbl + (size_t)BL * NPROJ;  // (48,256) zero-padded x_proj_w
    float* yg    = xi;               // reuse: xi dead after conv

    // 0) zero-pad x_proj_w so the x_proj GEMM is branch-free
    pad_xproj_kernel<<<(NPROJ_PAD * D_INNER) / 256, 256, 0, stream>>>(x_proj_w, wpad);

    // 1) in_proj: (BL,128)x(512,128)^T -> xi | z split at 256.
    //    16x64 tile/wave, K=128: 16 WMMAs per wave.
    {
        const int waves = (BL / 16) * (512 / 64);     // 1152*8 = 9216
        gemm_bf16_wmma<4, 4><<<waves / 8, 256, 0, stream>>>(
            x, in_proj, xi, z, BL / 16, 512 / 64, 512, 256, D_INNER, D_INNER);
    }
    // 2) depthwise conv 3x3 + SiLU
    conv_silu_kernel<<<(BL * D_INNER) / 256, 256, 0, stream>>>(xi, conv_w, conv_b, xs);
    // 3) x_proj: (BL,256)x(48,256)^T -> xdbl (store only n<40).
    //    16x48 tile/wave, K=256: 24 WMMAs per wave.
    {
        const int waves = (BL / 16) * 1;              // 1152
        gemm_bf16_wmma<8, 3><<<waves / 8, 256, 0, stream>>>(
            xs, wpad, xdbl, nullptr, BL / 16, 1, NPROJ, 1 << 30, NPROJ, 0);
    }
    // 4) dt projection + softplus
    delta_kernel<<<(BL * D_INNER) / 256, 256, 0, stream>>>(xdbl, dt_w, dt_b, delta);
    // 5) selective scan (serial over L, 512-way parallel, 16 states per lane-group)
    scan_kernel<<<128, 64, 0, stream>>>(xs, delta, xdbl, A_logs, Ds, ybuf);
    // 6) LayerNorm + SiLU gate
    ln_gate_kernel<<<BL, 256, 0, stream>>>(ybuf, z, ln_g, ln_b, yg);
    // 7) out_proj: (BL,256)x(128,256)^T -> d_out.
    //    16x64 tile/wave, K=256: 32 WMMAs per wave.
    {
        const int waves = (BL / 16) * (D_MODEL / 64); // 1152*2 = 2304
        gemm_bf16_wmma<8, 4><<<waves / 8, 256, 0, stream>>>(
            yg, out_proj, (float*)d_out, nullptr, BL / 16, D_MODEL / 64, D_MODEL,
            1 << 30, D_MODEL, 0);
    }
}
